// Seq2SeqBiLSTM_80101140070935
// MI455X (gfx1250) — compile-verified, matrix operand order checked by reading, not run on hardware
//
#include <hip/hip_runtime.h>

// ---------------------------------------------------------------------------
// Seq2Seq BiLSTM for MI455X (gfx1250): bf16 WMMA fast path, fp32 accumulate.
// V=10000 E=512 H=1024 D=2048 B=32 S=256
// ---------------------------------------------------------------------------

typedef __bf16 bf16_t;
typedef __attribute__((ext_vector_type(16))) __bf16 v16bf;
typedef __attribute__((ext_vector_type(8)))  __bf16 v8bf;
typedef __attribute__((ext_vector_type(8)))  float  v8f;

#define VOCAB 10000
#define EDIM  512
#define HDIM  1024
#define DDIM  2048
#define BATCH 32
#define SEQ   256

// ---- fragment load: 16-bit A/B operand, 16x32 (or 32x16) tile -------------
// Lane l (0..15) holds row/col (tile + (l&15)); lane half = l>>4 selects
// K ranges {half*8 .. half*8+7} and {16+half*8 .. 16+half*8+7}.
__device__ __forceinline__ v16bf load_frag(const bf16_t* __restrict__ rowptr,
                                           int k0, int half) {
  const v8bf lo = *reinterpret_cast<const v8bf*>(rowptr + k0 + half * 8);
  const v8bf hi = *reinterpret_cast<const v8bf*>(rowptr + k0 + 16 + half * 8);
  v16bf r;
#pragma unroll
  for (int i = 0; i < 8; ++i) { r[i] = lo[i]; r[i + 8] = hi[i]; }
  return r;
}

// ---- generic WMMA GEMM:  Out[32 x N] = A1[32 x K1] @ W1^T (+ A2 @ W2^T) + b
// A1 rows optionally gathered via token indices (embedding lookup fused).
// W matrices are [N x K] row-major (as in the reference), i.e. the natural
// K-contiguous B-operand for g = x @ W^T.
// Block = 64 threads (2 waves); wave w handles m-tile w, block.x = n-tile.
__global__ void __launch_bounds__(64)
wmma_gemm_bf16(const bf16_t* __restrict__ A1,
               const int* __restrict__ tokidx, int tok_stride, int tok_off,
               int lda1, int K1, const bf16_t* __restrict__ W1,
               const bf16_t* __restrict__ A2, int lda2, int K2,
               const bf16_t* __restrict__ W2,
               const float* __restrict__ bias,
               float* __restrict__ Out, long long out_row_stride) {
  const int lane = threadIdx.x & 31;
  const int m0   = (threadIdx.x >> 5) * 16;   // 0 or 16
  const int n0   = blockIdx.x * 16;
  const int r    = lane & 15;
  const int half = lane >> 4;

  // A-operand row pointer for this lane (row m0+r of the M=32 activations)
  const int m = m0 + r;
  const bf16_t* a1row =
      tokidx ? A1 + (size_t)tokidx[(size_t)m * tok_stride + tok_off] * lda1
             : A1 + (size_t)m * lda1;
  // B-operand "column" pointer = weight row n0+r (K-contiguous)
  const bf16_t* w1row = W1 + (size_t)(n0 + r) * K1;

  v8f acc = {};
#pragma unroll 4
  for (int k0 = 0; k0 < K1; k0 += 32) {
    v16bf a = load_frag(a1row, k0, half);
    v16bf b = load_frag(w1row, k0, half);
    acc = __builtin_amdgcn_wmma_f32_16x16x32_bf16(false, a, false, b,
                                                  (short)0, acc, false, false);
  }
  if (K2 > 0) {
    const bf16_t* a2row = A2 + (size_t)m * lda2;
    const bf16_t* w2row = W2 + (size_t)(n0 + r) * K2;
#pragma unroll 4
    for (int k0 = 0; k0 < K2; k0 += 32) {
      v16bf a = load_frag(a2row, k0, half);
      v16bf b = load_frag(w2row, k0, half);
      acc = __builtin_amdgcn_wmma_f32_16x16x32_bf16(false, a, false, b,
                                                    (short)0, acc, false, false);
    }
  }
  // C/D layout: vgpr j, lane l -> M = m0 + j + 8*(l>>4), N = n0 + (l&15)
  const int ncol = n0 + (lane & 15);
  const float bv = bias ? bias[ncol] : 0.0f;
#pragma unroll
  for (int j = 0; j < 8; ++j) {
    const int mrow = m0 + j + 8 * half;
    Out[(size_t)mrow * out_row_stride + ncol] = acc[j] + bv;
  }
}

// ---- LSTM pointwise: gates (B x 4Hx) + c -> new c, h (f32 + bf16) ----------
__global__ void lstm_pointwise(const float* __restrict__ G, int Hx,
                               const float* __restrict__ c_in,
                               float* __restrict__ c_out,
                               float* __restrict__ h_out,
                               bf16_t* __restrict__ h16_out) {
  const int idx = blockIdx.x * blockDim.x + threadIdx.x;  // b*Hx + j
  if (idx >= BATCH * Hx) return;
  const int b = idx / Hx, j = idx - b * Hx;
  const float* g = G + (size_t)b * 4 * Hx;
  const float gi = g[j], gf = g[Hx + j], gg = g[2 * Hx + j], go = g[3 * Hx + j];
  const float si = 1.0f / (1.0f + __expf(-gi));
  const float sf = 1.0f / (1.0f + __expf(-gf));
  const float so = 1.0f / (1.0f + __expf(-go));
  const float c = sf * c_in[idx] + si * tanhf(gg);
  const float h = so * tanhf(c);
  c_out[idx]   = c;
  h_out[idx]   = h;
  h16_out[idx] = (bf16_t)h;
}

// ---- batch-norm (stats over B=32) + ReLU; optional two-source concat -------
__global__ void bn_relu(const float* __restrict__ srcA,
                        const float* __restrict__ srcB, int splitD, int Dtot,
                        const float* __restrict__ gamma,
                        const float* __restrict__ beta,
                        float* __restrict__ out_f32,
                        bf16_t* __restrict__ out_bf16) {
  const int d = blockIdx.x * blockDim.x + threadIdx.x;
  if (d >= Dtot) return;
  float vals[BATCH];
  float s = 0.0f, s2 = 0.0f;
#pragma unroll
  for (int b = 0; b < BATCH; ++b) {
    const float v = (d < splitD) ? srcA[(size_t)b * splitD + d]
                                 : srcB[(size_t)b * splitD + (d - splitD)];
    vals[b] = v; s += v; s2 += v * v;
  }
  const float m    = s * (1.0f / BATCH);
  const float var  = s2 * (1.0f / BATCH) - m * m;
  const float gmul = gamma[d] * rsqrtf(var + 1e-5f);
  const float badd = beta[d];
#pragma unroll
  for (int b = 0; b < BATCH; ++b) {
    const float o = fmaxf((vals[b] - m) * gmul + badd, 0.0f);
    if (out_f32) out_f32[(size_t)b * Dtot + d] = o;
    out_bf16[(size_t)b * Dtot + d] = (bf16_t)o;
  }
}

// ---- per-row argmax over V (first-max tie-break like jnp.argmax) -----------
__global__ void argmax_rows(const float* __restrict__ logits,
                            long long row_stride, int Vn,
                            int* __restrict__ tok) {
  __shared__ float smax[256];
  __shared__ int   sidx[256];
  const int b = blockIdx.x;
  const float* row = logits + (size_t)b * row_stride;
  float best = -__builtin_inff();
  int   bi   = 0;
  for (int v = threadIdx.x; v < Vn; v += blockDim.x) {
    const float x = row[v];
    if (x > best) { best = x; bi = v; }
  }
  smax[threadIdx.x] = best; sidx[threadIdx.x] = bi;
  __syncthreads();
  for (int off = blockDim.x / 2; off > 0; off >>= 1) {
    if ((int)threadIdx.x < off) {
      const float o = smax[threadIdx.x + off];
      const int  oi = sidx[threadIdx.x + off];
      if (o > smax[threadIdx.x] ||
          (o == smax[threadIdx.x] && oi < sidx[threadIdx.x])) {
        smax[threadIdx.x] = o; sidx[threadIdx.x] = oi;
      }
    }
    __syncthreads();
  }
  if (threadIdx.x == 0) tok[b] = sidx[0];
}

// ---- helpers ---------------------------------------------------------------
__global__ void cvt_f32_bf16(const float* __restrict__ src,
                             bf16_t* __restrict__ dst, long long n) {
  long long i = (long long)blockIdx.x * blockDim.x + threadIdx.x;
  const long long stride = (long long)gridDim.x * blockDim.x;
  for (; i < n; i += stride) dst[i] = (bf16_t)src[i];
}

__global__ void fill_zero_f32(float* __restrict__ p, long long n) {
  long long i = (long long)blockIdx.x * blockDim.x + threadIdx.x;
  const long long stride = (long long)gridDim.x * blockDim.x;
  for (; i < n; i += stride) p[i] = 0.0f;
}

__global__ void init_tok(const int* __restrict__ x, int* __restrict__ tok) {
  if (threadIdx.x < BATCH) tok[threadIdx.x] = x[(size_t)threadIdx.x * SEQ];
}

__global__ void zero_first_step(float* __restrict__ out) {
  const long long i = (long long)blockIdx.x * blockDim.x + threadIdx.x;
  if (i < (long long)BATCH * VOCAB) {
    const long long b = i / VOCAB, v = i - b * VOCAB;
    out[(b * SEQ) * (long long)VOCAB + v] = 0.0f;
  }
}

// ---------------------------------------------------------------------------
extern "C" void kernel_launch(void* const* d_in, const int* in_sizes, int n_in,
                              void* d_out, int out_size, void* d_ws,
                              size_t ws_size, hipStream_t stream) {
  (void)in_sizes; (void)n_in; (void)out_size; (void)ws_size;

  const int*   x      = (const int*)  d_in[0];
  const float* emb    = (const float*)d_in[1];
  const float* Wih_f  = (const float*)d_in[2];
  const float* Whh_f  = (const float*)d_in[3];
  const float* b_f    = (const float*)d_in[4];
  const float* Wih_b  = (const float*)d_in[5];
  const float* Whh_b  = (const float*)d_in[6];
  const float* b_b    = (const float*)d_in[7];
  const float* Wih_d  = (const float*)d_in[8];
  const float* Whh_d  = (const float*)d_in[9];
  const float* b_d    = (const float*)d_in[10];
  const float* eg     = (const float*)d_in[11];
  const float* eb     = (const float*)d_in[12];
  const float* dg     = (const float*)d_in[13];
  const float* db     = (const float*)d_in[14];
  const float* fcW    = (const float*)d_in[15];
  const float* fcb    = (const float*)d_in[16];
  float* out = (float*)d_out;

  // ---- carve workspace (256B aligned) ----
  char* wp = (char*)d_ws;
  auto carve = [&](size_t bytes) -> char* {
    char* p = wp;
    wp += (bytes + 255) & ~(size_t)255;
    return p;
  };
  bf16_t* emb16 = (bf16_t*)carve((size_t)VOCAB * EDIM * 2);
  bf16_t* Wfi16 = (bf16_t*)carve((size_t)4 * HDIM * EDIM * 2);
  bf16_t* Wfh16 = (bf16_t*)carve((size_t)4 * HDIM * HDIM * 2);
  bf16_t* Wbi16 = (bf16_t*)carve((size_t)4 * HDIM * EDIM * 2);
  bf16_t* Wbh16 = (bf16_t*)carve((size_t)4 * HDIM * HDIM * 2);
  bf16_t* Wdi16 = (bf16_t*)carve((size_t)4 * DDIM * EDIM * 2);
  bf16_t* Wdh16 = (bf16_t*)carve((size_t)4 * DDIM * DDIM * 2);
  bf16_t* fc16  = (bf16_t*)carve((size_t)VOCAB * DDIM * 2);

  char* state_begin = wp;
  float*  h_f   = (float*) carve((size_t)BATCH * HDIM * 4);
  float*  c_f   = (float*) carve((size_t)BATCH * HDIM * 4);
  float*  h_b   = (float*) carve((size_t)BATCH * HDIM * 4);
  float*  c_b   = (float*) carve((size_t)BATCH * HDIM * 4);
  bf16_t* hf16  = (bf16_t*)carve((size_t)BATCH * HDIM * 2);
  bf16_t* hb16  = (bf16_t*)carve((size_t)BATCH * HDIM * 2);
  float*  dech  = (float*) carve((size_t)BATCH * DDIM * 4);
  float*  decc  = (float*) carve((size_t)BATCH * DDIM * 4);
  bf16_t* dech16= (bf16_t*)carve((size_t)BATCH * DDIM * 2);
  bf16_t* bno16 = (bf16_t*)carve((size_t)BATCH * DDIM * 2);
  float*  gates = (float*) carve((size_t)BATCH * 4 * DDIM * 4);
  int*    tok   = (int*)   carve((size_t)BATCH * 4);
  char* state_end = wp;

  // ---- one-time (per launch) conversions + init ----
  auto cvt = [&](const float* s, bf16_t* d, long long n) {
    int grid = (int)((n + 255) / 256); if (grid > 4096) grid = 4096;
    hipLaunchKernelGGL(cvt_f32_bf16, dim3(grid), dim3(256), 0, stream, s, d, n);
  };
  cvt(emb,   emb16, (long long)VOCAB * EDIM);
  cvt(Wih_f, Wfi16, (long long)4 * HDIM * EDIM);
  cvt(Whh_f, Wfh16, (long long)4 * HDIM * HDIM);
  cvt(Wih_b, Wbi16, (long long)4 * HDIM * EDIM);
  cvt(Whh_b, Wbh16, (long long)4 * HDIM * HDIM);
  cvt(Wih_d, Wdi16, (long long)4 * DDIM * EDIM);
  cvt(Whh_d, Wdh16, (long long)4 * DDIM * DDIM);
  cvt(fcW,   fc16,  (long long)VOCAB * DDIM);

  {
    const long long nf = (long long)(state_end - state_begin) / 4;
    int grid = (int)((nf + 255) / 256); if (grid > 4096) grid = 4096;
    hipLaunchKernelGGL(fill_zero_f32, dim3(grid), dim3(256), 0, stream,
                       (float*)state_begin, nf);
  }
  hipLaunchKernelGGL(zero_first_step,
                     dim3((BATCH * VOCAB + 255) / 256), dim3(256), 0, stream,
                     out);

  // ---- encoder: 256 steps, fwd + bwd ----
  const dim3 gemmBlk(64);
  for (int t = 0; t < SEQ; ++t) {
    // forward direction, input token x[b][t]
    hipLaunchKernelGGL(wmma_gemm_bf16, dim3(4 * HDIM / 16), gemmBlk, 0, stream,
                       emb16, x, SEQ, t, EDIM, EDIM, Wfi16,
                       hf16, HDIM, HDIM, Wfh16, b_f,
                       gates, (long long)(4 * HDIM));
    hipLaunchKernelGGL(lstm_pointwise, dim3((BATCH * HDIM + 255) / 256),
                       dim3(256), 0, stream, gates, HDIM, c_f, c_f, h_f, hf16);
    // backward direction, input token x[b][S-1-t]
    hipLaunchKernelGGL(wmma_gemm_bf16, dim3(4 * HDIM / 16), gemmBlk, 0, stream,
                       emb16, x, SEQ, SEQ - 1 - t, EDIM, EDIM, Wbi16,
                       hb16, HDIM, HDIM, Wbh16, b_b,
                       gates, (long long)(4 * HDIM));
    hipLaunchKernelGGL(lstm_pointwise, dim3((BATCH * HDIM + 255) / 256),
                       dim3(256), 0, stream, gates, HDIM, c_b, c_b, h_b, hb16);
  }

  // hn = relu(BN(concat(h_f, h_b))) -> decoder initial h (f32 + bf16)
  hipLaunchKernelGGL(bn_relu, dim3((DDIM + 255) / 256), dim3(256), 0, stream,
                     h_f, h_b, HDIM, DDIM, eg, eb, dech, dech16);
  hipLaunchKernelGGL(init_tok, dim3(1), dim3(32), 0, stream, x, tok);
  // decc already zeroed by state fill

  // ---- decoder: 255 steps ----
  for (int t = 0; t < SEQ - 1; ++t) {
    // gates = emb[tok] @ Wih_d^T + h @ Whh_d^T + b_d
    hipLaunchKernelGGL(wmma_gemm_bf16, dim3(4 * DDIM / 16), gemmBlk, 0, stream,
                       emb16, tok, 1, 0, EDIM, EDIM, Wdi16,
                       dech16, DDIM, DDIM, Wdh16, b_d,
                       gates, (long long)(4 * DDIM));
    hipLaunchKernelGGL(lstm_pointwise, dim3((BATCH * DDIM + 255) / 256),
                       dim3(256), 0, stream, gates, DDIM, decc, decc, dech,
                       dech16);
    // out = relu(BN(h)) (bf16 for logits GEMM)
    hipLaunchKernelGGL(bn_relu, dim3((DDIM + 255) / 256), dim3(256), 0, stream,
                       dech, dech, DDIM, DDIM, dg, db, (float*)nullptr, bno16);
    // logits -> d_out[:, t+1, :], then argmax -> next token
    float* logits = out + (size_t)(t + 1) * VOCAB;
    hipLaunchKernelGGL(wmma_gemm_bf16, dim3(VOCAB / 16), gemmBlk, 0, stream,
                       bno16, (const int*)nullptr, 0, 0, DDIM, DDIM, fc16,
                       (const bf16_t*)nullptr, 0, 0, (const bf16_t*)nullptr,
                       fcb, logits, (long long)SEQ * VOCAB);
    hipLaunchKernelGGL(argmax_rows, dim3(BATCH), dim3(256), 0, stream,
                       logits, (long long)SEQ * VOCAB, VOCAB, tok);
  }
}
